// DMSP_51256139710777
// MI455X (gfx1250) — matrix operand order repaired and checked by reading, not compiled, not run on hardware
//
#include <hip/hip_runtime.h>
#include <math.h>

typedef __attribute__((ext_vector_type(16))) _Float16 v16h;
typedef __attribute__((ext_vector_type(8)))  _Float16 v8h;
typedef __attribute__((ext_vector_type(8)))  float    v8f;

#define NTB   1000
#define BSZ   2
#define HIDN  1024
#define NMOD  64
#define NBND  65
#define BLK   192
#define NPAD  208
#define LTOT  192000
#define TWO_PI 6.283185307179586f
#define LOG10F_C 2.302585092994046f

// ---------------------------------------------------------------------------
// Kernel 1: convert hidden f32 -> f16, build transposed zero-padded weight
// wt16 is (NPAD x HIDN) row-major: rows 0..63 = W_fm cols, 64..127 = W_am cols,
// 128..192 = W_noise cols, 193..207 = zero padding.
// ---------------------------------------------------------------------------
__global__ void dmsp_pack(const float* __restrict__ hidden,
                          const float* __restrict__ W_fm,
                          const float* __restrict__ W_am,
                          const float* __restrict__ W_noise,
                          _Float16* __restrict__ hid16,
                          _Float16* __restrict__ wt16) {
    const int NH = BSZ * NTB * HIDN;          // 2,048,000
    const int NW = NPAD * HIDN;               // 212,992
    int i = blockIdx.x * blockDim.x + threadIdx.x;
    if (i < NH) {
        hid16[i] = (_Float16)hidden[i];
    } else if (i < NH + NW) {
        int j = i - NH;
        int n = j / HIDN;
        int k = j - n * HIDN;
        float v = 0.0f;
        if (n < 64)        v = W_fm[k * 64 + n];
        else if (n < 128)  v = W_am[k * 64 + (n - 64)];
        else if (n < 193)  v = W_noise[k * 65 + (n - 128)];
        wt16[n * HIDN + k] = (_Float16)v;
    }
}

// ---------------------------------------------------------------------------
// Kernel 2: WMMA f16 GEMM, C(2000 x NPAD) = A(2000 x 1024) * BT^T
// One wave per 16x16 C tile; BT is (NPAD x 1024) so B fragments are contiguous.
// A fragment (16x32 f16): lane = half*16 + m; elems 0..15 -> K=k0+half*16..
//   (16 contiguous K per lane-half, matching B)
// B fragment (32x16 f16): lane = half*16 + n; elems 0..15 -> K=k0+half*16..
//   (16 contiguous K per lane-half, same order as A)
// C/D: VGPR r, lane half -> row r + 8*half, col lane&15.
// ---------------------------------------------------------------------------
__global__ __launch_bounds__(32) void dmsp_gemm_wmma(
        const _Float16* __restrict__ A,
        const _Float16* __restrict__ BT,
        float* __restrict__ C) {
    int tm = blockIdx.x;              // 0..124
    int tn = blockIdx.y;              // 0..12
    int lane = threadIdx.x;           // 0..31
    int rc   = lane & 15;
    int half = lane >> 4;
    const _Float16* arow = A  + (size_t)(tm * 16 + rc) * HIDN;
    const _Float16* brow = BT + (size_t)(tn * 16 + rc) * HIDN;
    v8f c = {};
    for (int k0 = 0; k0 < HIDN; k0 += 32) {
        v8h alo = *(const v8h*)(arow + k0 + half * 16);
        v8h ahi = *(const v8h*)(arow + k0 + half * 16 + 8);
        v16h a, b;
#pragma unroll
        for (int i = 0; i < 8; ++i) { a[i] = alo[i]; a[i + 8] = ahi[i]; }
        b = *(const v16h*)(brow + k0 + half * 16);
        c = __builtin_amdgcn_wmma_f32_16x16x32_f16(
                false, a, false, b, (short)0, c, false, false);
    }
    float* crow = C + (size_t)(tm * 16 + half * 8) * NPAD + tn * 16 + rc;
#pragma unroll
    for (int r = 0; r < 8; ++r) crow[r * NPAD] = c[r];
}

// ---------------------------------------------------------------------------
// Kernel 3: per-(b,t,m) mode params -> freq_m, coef_m (written to d_out)
// ---------------------------------------------------------------------------
__global__ void dmsp_modes(const float* __restrict__ lin,
                           const float* __restrict__ mode_freq,
                           const float* __restrict__ mode_coef,
                           const float* __restrict__ times,
                           const float* __restrict__ alpha,
                           const float* __restrict__ omega,
                           const float* __restrict__ b_fm,
                           const float* __restrict__ b_am,
                           float* __restrict__ out_freq,
                           float* __restrict__ out_coef) {
    int idx = blockIdx.x * blockDim.x + threadIdx.x;
    if (idx >= BSZ * NTB * NMOD) return;
    int b = idx / (NTB * NMOD);
    int r = idx - b * (NTB * NMOD);
    int t = r / NMOD;
    int m = r - t * NMOD;
    int row = b * NTB + t;

    float xf = lin[(size_t)row * NPAD + m]      + b_fm[m];
    float xa = lin[(size_t)row * NPAD + 64 + m] + b_am[m];
    float delta = tanhf(xf);
    float sig   = (xa > 20.0f) ? xa : log1pf(expf(xa));

    float freq = mode_freq[idx] * sqrtf(1.0f + 0.01f * alpha[b])
               + 0.01f * omega[b] * delta;
    float coef = mode_coef[b * NMOD + m] * expf(-sig * times[row]);
    float fhz  = freq * (48000.0f / TWO_PI);
    float mask = (fhz < 24000.0f) ? 1.0f : 0.0f;
    coef *= (mask + 1e-4f);

    out_freq[idx] = freq;
    out_coef[idx] = coef;
}

// ---------------------------------------------------------------------------
// Kernel 4: block-level exclusive prefix of phase: P[b][n][m] = 192*sum_{j<n} f
// 128 independent sequences of length 1000.
// ---------------------------------------------------------------------------
__global__ __launch_bounds__(128) void dmsp_scan(const float* __restrict__ freq,
                                                 float* __restrict__ P) {
    int id = threadIdx.x;                 // 0..127
    int b = id >> 6;
    int m = id & 63;
    float acc = 0.0f;
    for (int n = 0; n < NTB; ++n) {
        size_t o = ((size_t)(b * NTB + n)) * NMOD + m;
        P[o] = acc;
        acc += 192.0f * freq[o];
    }
}

// ---------------------------------------------------------------------------
// Kernel 5: fused noise-filter + modal synthesis, one block per (b, block).
//   amp[j]  = scale_function(lin_noise + b_noise - 5)
//   ir0[j]  = irfft(amp)[j] (65-term cosine sum), tap[j] = ir0[j]*hann[(j+64)&127]
//   delays: j<64 -> d=j ; j>=64 -> d=j+64       (roll/pad/roll collapsed)
//   nf[t]   = sum_j tap[j] * n2[t-d]  (causal)
//   sig[t]  = sum_m coef[m]*sin(P[m] + (t+1)*freq[m]) + nf[t]
// ---------------------------------------------------------------------------
__global__ __launch_bounds__(BLK) void dmsp_synth(
        const float* __restrict__ lin,
        const float* __restrict__ b_noise,
        const float* __restrict__ noise,
        const float* __restrict__ freq,
        const float* __restrict__ coef,
        const float* __restrict__ P,
        float* __restrict__ out_sig) {
    __shared__ float s_amp[NBND];
    __shared__ float s_tap[128];
    __shared__ float s_n2[BLK];
    __shared__ float s_f[NMOD];
    __shared__ float s_p[NMOD];
    __shared__ float s_c[NMOD];

    int blk = blockIdx.x;             // 0..1999
    int b = blk / NTB;
    int n = blk - b * NTB;
    int t = threadIdx.x;              // 0..191
    int row = b * NTB + n;

    if (t < NBND) {
        float x = lin[(size_t)row * NPAD + 128 + t] + b_noise[t] - 5.0f;
        float s = 1.0f / (1.0f + expf(-x));
        s_amp[t] = 2.0f * expf(LOG10F_C * logf(s)) + 1e-7f;   // sigmoid^ln(10)
    }
    if (t < NMOD) {
        size_t o = (size_t)row * NMOD + t;
        s_f[t] = freq[o];
        s_p[t] = P[o];
        s_c[t] = coef[o];
    }
    s_n2[t] = noise[(size_t)row * BLK + t] * 2.0f - 1.0f;
    __syncthreads();

    if (t < 128) {
        float acc = 0.0f;
        float w0 = (TWO_PI / 128.0f) * (float)t;
#pragma unroll 8
        for (int k = 0; k <= 64; ++k) {
            float wgt = (k == 0 || k == 64) ? 1.0f : 2.0f;
            acc += wgt * s_amp[k] * cosf(w0 * (float)k);
        }
        acc *= (1.0f / 128.0f);
        int hw = (t + 64) & 127;
        float win = 0.5f * (1.0f - cosf((TWO_PI / 128.0f) * (float)hw));
        s_tap[t] = acc * win;
    }
    __syncthreads();

    // causal truncated convolution (taps at delays 0..63 and 128..191)
    float nf = 0.0f;
#pragma unroll 4
    for (int j = 0; j < 64; ++j) {
        int s = t - j;
        if (s >= 0) nf += s_tap[j] * s_n2[s];
    }
#pragma unroll 4
    for (int j = 64; j < 128; ++j) {
        int s = t - (j + 64);
        if (s >= 0) nf += s_tap[j] * s_n2[s];
    }

    // modal sinusoid bank
    float h = 0.0f;
    float kp1 = (float)(t + 1);
#pragma unroll 8
    for (int m = 0; m < NMOD; ++m)
        h += s_c[m] * sinf(s_p[m] + kp1 * s_f[m]);

    out_sig[(size_t)b * LTOT + n * BLK + t] = h + nf;
}

// ---------------------------------------------------------------------------
extern "C" void kernel_launch(void* const* d_in, const int* in_sizes, int n_in,
                              void* d_out, int out_size, void* d_ws, size_t ws_size,
                              hipStream_t stream) {
    const float* hidden    = (const float*)d_in[0];
    const float* mode_freq = (const float*)d_in[1];
    const float* mode_coef = (const float*)d_in[2];
    const float* times     = (const float*)d_in[3];
    const float* alpha     = (const float*)d_in[4];
    const float* omega     = (const float*)d_in[5];
    const float* noise     = (const float*)d_in[6];
    const float* W_fm      = (const float*)d_in[7];
    const float* b_fm      = (const float*)d_in[8];
    const float* W_am      = (const float*)d_in[9];
    const float* b_am      = (const float*)d_in[10];
    const float* W_noise   = (const float*)d_in[11];
    const float* b_noise   = (const float*)d_in[12];
    // d_in[13] = lengths (== 192000 == NT*BLOCK, crop is a no-op)

    char* ws = (char*)d_ws;
    _Float16* hid16 = (_Float16*)ws;                                   // 4,096,000 B
    _Float16* wt16  = (_Float16*)(ws + 4096000);                       //   425,984 B
    float*    lin   = (float*)(ws + 4096000 + 425984);                 // 1,664,000 B
    float*    P     = (float*)(ws + 4096000 + 425984 + 1664000);       //   512,000 B

    float* out_sig  = (float*)d_out;               // (2,192000)
    float* out_freq = out_sig + BSZ * LTOT;        // (2,1000,64)
    float* out_coef = out_freq + BSZ * NTB * NMOD; // (2,1000,64)

    // 1) pack: f16 conversion + transposed padded weight
    {
        int total = BSZ * NTB * HIDN + NPAD * HIDN;   // 2,260,992
        int grid = (total + 255) / 256;
        dmsp_pack<<<grid, 256, 0, stream>>>(hidden, W_fm, W_am, W_noise,
                                            hid16, wt16);
    }
    // 2) WMMA GEMM -> lin (2000 x 208)
    dmsp_gemm_wmma<<<dim3(125, 13), 32, 0, stream>>>(hid16, wt16, lin);

    // 3) mode params -> freq_m / coef_m in d_out
    {
        int total = BSZ * NTB * NMOD;                 // 128,000
        dmsp_modes<<<(total + 255) / 256, 256, 0, stream>>>(
            lin, mode_freq, mode_coef, times, alpha, omega, b_fm, b_am,
            out_freq, out_coef);
    }
    // 4) per-(b,m) block phase prefix
    dmsp_scan<<<1, 128, 0, stream>>>(out_freq, P);

    // 5) fused IR + conv + sinusoid bank -> signal
    dmsp_synth<<<BSZ * NTB, BLK, 0, stream>>>(lin, b_noise, noise,
                                              out_freq, out_coef, P, out_sig);
}
